// AttnGCN_70274254897747
// MI455X (gfx1250) — compile-verified, matrix operand order checked by reading, not run on hardware
//
#include <hip/hip_runtime.h>
#include <hip/hip_bf16.h>
#include <math.h>

// GATv2 x2 + BN + dropout + FC for MI455X (gfx1250, wave32, WMMA).
// Dense transforms: f16 WMMA 16x16x32, f32 accumulate, LDS-tiled with
// fragment-friendly layouts (two ds_load_b128 per fragment).
// Edge phase: wave-per-edge gather/reduce + f32 global atomics.

#define HC   256      // H*C channels
#define NCLS 26       // in/out classes

typedef __attribute__((ext_vector_type(16))) _Float16 v16h;
typedef __attribute__((ext_vector_type(8)))  _Float16 v8h;
typedef __attribute__((ext_vector_type(4)))  _Float16 v4h;
typedef __attribute__((ext_vector_type(8)))  float    v8f;
typedef __attribute__((ext_vector_type(4)))  float    v4f;

static __device__ __forceinline__ void atomicMaxF(float* addr, float val) {
  int* ai = (int*)addr;
  int old = __float_as_int(*addr);
  while (__int_as_float(old) < val) {
    int assumed = old;
    old = atomicCAS(ai, assumed, __float_as_int(val));
    if (old == assumed) break;
  }
}

// ---------------------------------------------------------------------------
// WMMA GEMM: C[M x NCOL] = A[M x K] @ W[K x NCOL] + bias
// Block tile 64x64, 256 threads = 8 waves (2x4); each wave owns two 16x16
// accumulators stacked in M -> 2 wmma per K-step per wave.
// LDS: A row-major (rows of 32 halves), W transposed (Bst[n][k]) so every
// fragment is two aligned 16-byte LDS reads (ds_load_b128).
// VEC=true: K%32==0 && NCOL%64==0 (float4 global staging, row-clamp only).
// ---------------------------------------------------------------------------
template <bool VEC>
__global__ __launch_bounds__(256) void k_gemm_wmma(
    const float* __restrict__ A, const float* __restrict__ W,
    const float* __restrict__ bias, float* __restrict__ C,
    int M, int K, int NCOL)
{
  const int tid  = threadIdx.x;
  const int lane = tid & 31;
  const int wave = tid >> 5;
  const int waveRow = wave >> 2;   // 0..1  (M half of 64)
  const int waveCol = wave & 3;    // 0..3  (N tile of 16)
  const int m0 = blockIdx.x * 64;
  const int n0 = blockIdx.y * 64;

  __shared__ __align__(16) _Float16 As [64][40]; // 64 rows x 32 k (+pad, 16B-aligned rows)
  __shared__ __align__(16) _Float16 Bst[64][40]; // 64 n    x 32 k (transposed W tile)

  v8f acc0 = {}, acc1 = {};

  const int hs = lane >> 4;     // half-wave: selects K sub-stripe
  const int rc = lane & 15;     // row (A) / col (B) within 16x16 tile

  for (int k0 = 0; k0 < K; k0 += 32) {
    if (VEC) {
      // ---- A tile: 64x32 f32, 512 float4 loads (2/thread), row clamp only
      #pragma unroll
      for (int i = 0; i < 2; ++i) {
        int q   = i * 256 + tid;          // 0..511
        int r   = q >> 3;                 // 0..63
        int kq  = (q & 7) * 4;            // 0,4,...,28
        int gr  = m0 + r; gr = gr < M ? gr : M - 1;
        v4f v = *(const v4f*)(A + (long long)gr * K + k0 + kq);
        v4h h; h[0]=(_Float16)v[0]; h[1]=(_Float16)v[1]; h[2]=(_Float16)v[2]; h[3]=(_Float16)v[3];
        *(v4h*)&As[r][kq] = h;
      }
      // ---- W tile: 32x64 f32, coalesced float4 read, transposed b16 writes
      #pragma unroll
      for (int i = 0; i < 2; ++i) {
        int q  = i * 256 + tid;           // 0..511
        int kr = q >> 4;                  // 0..31
        int nq = (q & 15) * 4;            // 0,4,...,60
        v4f v = *(const v4f*)(W + (long long)(k0 + kr) * NCOL + n0 + nq);
        Bst[nq + 0][kr] = (_Float16)v[0];
        Bst[nq + 1][kr] = (_Float16)v[1];
        Bst[nq + 2][kr] = (_Float16)v[2];
        Bst[nq + 3][kr] = (_Float16)v[3];
      }
    } else {
      // ---- generic path (K=26 / NCOL=26): clamped unconditional loads + select
      #pragma unroll
      for (int i = 0; i < 8; ++i) {
        int idx = i * 256 + tid;          // 0..2047  (64x32)
        int r = idx >> 5, kk = idx & 31;
        int gr = m0 + r, gk = k0 + kk;
        bool ok = (gr < M) && (gk < K);
        int cr = gr < M ? gr : M - 1;
        int ck = gk < K ? gk : K - 1;
        float v = A[(long long)cr * K + ck];
        As[r][kk] = (_Float16)(ok ? v : 0.0f);
      }
      #pragma unroll
      for (int i = 0; i < 8; ++i) {
        int idx = i * 256 + tid;          // 0..2047  (32x64)
        int kr = idx >> 6, nn = idx & 63;
        int gk = k0 + kr, gn = n0 + nn;
        bool ok = (gk < K) && (gn < NCOL);
        int ck = gk < K ? gk : K - 1;
        int cn = gn < NCOL ? gn : NCOL - 1;
        float v = W[(long long)ck * NCOL + cn];
        Bst[nn][kr] = (_Float16)(ok ? v : 0.0f);
      }
    }
    __syncthreads();

    // ---- fragments: two aligned 16B LDS reads each (K runs [8hs..] & [16+8hs..])
    const int ar0 = waveRow * 32 + rc;
    const int ar1 = ar0 + 16;
    const int bn  = waveCol * 16 + rc;
    v8h a0lo = *(const v8h*)&As[ar0][hs * 8];
    v8h a0hi = *(const v8h*)&As[ar0][16 + hs * 8];
    v8h a1lo = *(const v8h*)&As[ar1][hs * 8];
    v8h a1hi = *(const v8h*)&As[ar1][16 + hs * 8];
    v8h blo  = *(const v8h*)&Bst[bn][hs * 8];
    v8h bhi  = *(const v8h*)&Bst[bn][16 + hs * 8];
    v16h af0 = __builtin_shufflevector(a0lo, a0hi, 0,1,2,3,4,5,6,7,8,9,10,11,12,13,14,15);
    v16h af1 = __builtin_shufflevector(a1lo, a1hi, 0,1,2,3,4,5,6,7,8,9,10,11,12,13,14,15);
    v16h bf  = __builtin_shufflevector(blo,  bhi,  0,1,2,3,4,5,6,7,8,9,10,11,12,13,14,15);

    acc0 = __builtin_amdgcn_wmma_f32_16x16x32_f16(false, af0, false, bf, (short)0, acc0, false, false);
    acc1 = __builtin_amdgcn_wmma_f32_16x16x32_f16(false, af1, false, bf, (short)0, acc1, false, false);
    __syncthreads();
  }

  // ---- store per C layout: n = lane&15, m = (lane>>4)*8 + r
  const int gn = n0 + waveCol * 16 + rc;
  if (gn < NCOL) {
    const float b = bias[gn];
    #pragma unroll
    for (int r = 0; r < 8; ++r) {
      int gm = m0 + waveRow * 32 + hs * 8 + r;
      if (gm < M) C[(long long)gm * NCOL + gn] = acc0[r] + b;
      gm += 16;
      if (gm < M) C[(long long)gm * NCOL + gn] = acc1[r] + b;
    }
  }
}

// ---------------------------------------------------------------------------
// Edge kernels: one wave32 per edge; lane owns 8 channels; lanes 0-15 = head0,
// lanes 16-31 = head1 (att is flat HC so att index == channel index).
// ---------------------------------------------------------------------------
template <bool SELF>
__global__ __launch_bounds__(256) void k_edge_alpha(
    const long long* __restrict__ srcIdx, const long long* __restrict__ dstIdx,
    const float* __restrict__ ew, const float* __restrict__ lattr,
    const float* __restrict__ xl, const float* __restrict__ xr,
    const float* __restrict__ We, const float* __restrict__ att,
    float* __restrict__ alpha, long long nE)
{
  const int lane = threadIdx.x & 31;
  long long e = (long long)blockIdx.x * (blockDim.x >> 5) + (threadIdx.x >> 5);
  if (e >= nE) return;
  long long s, d; float w;
  if (SELF) { s = e; d = e; w = lattr[e]; }
  else      { s = srcIdx[e]; d = dstIdx[e]; w = ew[e]; }
  const float* __restrict__ xls = xl + s * HC;
  const float* __restrict__ xrd = xr + d * HC;
  __builtin_prefetch(xls, 0, 1);
  __builtin_prefetch(xrd, 0, 1);
  float part = 0.0f;
  const int cbase = lane * 8;
  v4f xa0 = *(const v4f*)(xls + cbase),  xa1 = *(const v4f*)(xls + cbase + 4);
  v4f xb0 = *(const v4f*)(xrd + cbase),  xb1 = *(const v4f*)(xrd + cbase + 4);
  v4f we0 = *(const v4f*)(We  + cbase),  we1 = *(const v4f*)(We  + cbase + 4);
  v4f at0 = *(const v4f*)(att + cbase),  at1 = *(const v4f*)(att + cbase + 4);
  #pragma unroll
  for (int j = 0; j < 4; ++j) {
    float m = xa0[j] + xb0[j] + w * we0[j];
    m = (m > 0.0f) ? m : 0.2f * m;           // leaky_relu 0.2
    part += m * at0[j];
    float m2 = xa1[j] + xb1[j] + w * we1[j];
    m2 = (m2 > 0.0f) ? m2 : 0.2f * m2;
    part += m2 * at1[j];
  }
  // reduce within each 16-lane half (one half per head)
  #pragma unroll
  for (int off = 8; off >= 1; off >>= 1)
    part += __shfl_xor(part, off, 32);
  if ((lane & 15) == 0) alpha[e * 2 + (lane >> 4)] = part;
}

template <bool SELF>
__global__ void k_alpha_max(const long long* __restrict__ dstIdx,
                            const float* __restrict__ alpha,
                            float* __restrict__ nmax, long long nE)
{
  long long i = (long long)blockIdx.x * blockDim.x + threadIdx.x;
  if (i >= nE * 2) return;
  long long e = i >> 1; int h = (int)(i & 1);
  long long d = SELF ? e : dstIdx[e];
  atomicMaxF(&nmax[d * 2 + h], alpha[i]);
}

template <bool SELF>
__global__ void k_alpha_exp(const long long* __restrict__ dstIdx,
                            float* __restrict__ alpha,
                            const float* __restrict__ nmax,
                            float* __restrict__ nden, long long nE)
{
  long long i = (long long)blockIdx.x * blockDim.x + threadIdx.x;
  if (i >= nE * 2) return;
  long long e = i >> 1; int h = (int)(i & 1);
  long long d = SELF ? e : dstIdx[e];
  float a = __expf(alpha[i] - nmax[d * 2 + h]);
  alpha[i] = a;
  atomicAdd(&nden[d * 2 + h], a);
}

template <bool SELF>
__global__ __launch_bounds__(256) void k_scatter(
    const long long* __restrict__ srcIdx, const long long* __restrict__ dstIdx,
    const float* __restrict__ alpha, const float* __restrict__ nden,
    const float* __restrict__ xl, float* __restrict__ out, long long nE)
{
  const int lane = threadIdx.x & 31;
  long long e = (long long)blockIdx.x * (blockDim.x >> 5) + (threadIdx.x >> 5);
  if (e >= nE) return;
  long long s, d;
  if (SELF) { s = e; d = e; }
  else      { s = srcIdx[e]; d = dstIdx[e]; }
  const int head = lane >> 4;
  const float coef = alpha[e * 2 + head] / (nden[d * 2 + head] + 1e-16f);
  const float* __restrict__ xls = xl + s * HC;
  float* __restrict__ od = out + d * HC;
  __builtin_prefetch(xls, 0, 1);
  const int cbase = lane * 8;
  v4f x0 = *(const v4f*)(xls + cbase);
  v4f x1 = *(const v4f*)(xls + cbase + 4);
  #pragma unroll
  for (int j = 0; j < 4; ++j) {
    atomicAdd(&od[cbase + j],     coef * x0[j]);
    atomicAdd(&od[cbase + 4 + j], coef * x1[j]);
  }
}

// ---------------------------------------------------------------------------
// Small elementwise / reduction kernels
// ---------------------------------------------------------------------------
__global__ void k_fill_bias(float* __restrict__ out, const float* __restrict__ b,
                            long long total)
{
  long long i = (long long)blockIdx.x * blockDim.x + threadIdx.x;
  if (i < total) out[i] = b[i & (HC - 1)];
}

__global__ void k_init_softmax(float* __restrict__ nmax, float* __restrict__ nden,
                               long long total)
{
  long long i = (long long)blockIdx.x * blockDim.x + threadIdx.x;
  if (i < total) { nmax[i] = -INFINITY; nden[i] = 0.0f; }
}

__global__ void k_zero2(float* __restrict__ a, float* __restrict__ b, long long total)
{
  long long i = (long long)blockIdx.x * blockDim.x + threadIdx.x;
  if (i < total) { a[i] = 0.0f; b[i] = 0.0f; }
}

__global__ void k_deg_sew(const long long* __restrict__ dstIdx,
                          const float* __restrict__ ew,
                          float* __restrict__ deg, float* __restrict__ sew,
                          long long nE)
{
  long long i = (long long)blockIdx.x * blockDim.x + threadIdx.x;
  if (i >= nE) return;
  long long d = dstIdx[i];
  atomicAdd(&deg[d], 1.0f);
  atomicAdd(&sew[d], ew[i]);
}

__global__ void k_loop_attr(const float* __restrict__ deg, float* __restrict__ sew,
                            long long n)
{
  long long i = (long long)blockIdx.x * blockDim.x + threadIdx.x;
  if (i < n) sew[i] = sew[i] / fmaxf(deg[i], 1.0f);
}

__global__ __launch_bounds__(256) void k_bn_stats(
    const float* __restrict__ h, float* __restrict__ mu,
    float* __restrict__ rsig, int n)
{
  const int c = blockIdx.x;  // channel 0..255
  float s = 0.0f, sq = 0.0f;
  for (int r = threadIdx.x; r < n; r += 256) {
    float v = h[(long long)r * HC + c];
    s += v; sq += v * v;
  }
  __shared__ float ss[256], sqs[256];
  ss[threadIdx.x] = s; sqs[threadIdx.x] = sq;
  __syncthreads();
  for (int st = 128; st > 0; st >>= 1) {
    if (threadIdx.x < st) {
      ss[threadIdx.x]  += ss[threadIdx.x + st];
      sqs[threadIdx.x] += sqs[threadIdx.x + st];
    }
    __syncthreads();
  }
  if (threadIdx.x == 0) {
    float m = ss[0] / n;
    float v = sqs[0] / n - m * m;   // biased variance (BatchNorm1d training)
    mu[c]   = m;
    rsig[c] = rsqrtf(v + 1e-5f);
  }
}

__global__ void k_bn_act_drop(float* __restrict__ h,
                              const float* __restrict__ mu,
                              const float* __restrict__ rsig,
                              const float* __restrict__ gamma,
                              const float* __restrict__ beta,
                              unsigned seed, long long total)
{
  long long i = (long long)blockIdx.x * blockDim.x + threadIdx.x;
  if (i >= total) return;
  const int c = (int)(i & (HC - 1));
  float v = gamma[c] * (h[i] - mu[c]) * rsig[c] + beta[c];
  v = (v > 0.0f) ? v : 0.01f * v;            // leaky_relu 0.01
  // deterministic counter-based dropout, keep prob 0.8
  unsigned x = (unsigned)i;
  unsigned y = (unsigned)(i >> 32) ^ seed;
  unsigned hsh = x * 0x9E3779B9u;
  hsh ^= y + 0x7F4A7C15u + (hsh << 6) + (hsh >> 2);
  hsh *= 0x85EBCA6Bu; hsh ^= hsh >> 13;
  hsh *= 0xC2B2AE35u; hsh ^= hsh >> 16;
  h[i] = ((hsh & 0xFFFFFFu) < 13421773u /*0.8*2^24*/) ? v * 1.25f : 0.0f;
}

// ---------------------------------------------------------------------------
// Host-side orchestration
// ---------------------------------------------------------------------------
extern "C" void kernel_launch(void* const* d_in, const int* in_sizes, int n_in,
                              void* d_out, int out_size, void* d_ws, size_t ws_size,
                              hipStream_t stream) {
  (void)n_in; (void)out_size; (void)ws_size;
  // Input order: x_input, edge_weight, params{Wl1,bl1,Wr1,br1,We1,att1,bo1,
  //   Wl2,bl2,Wr2,br2,We2,att2,bo2,g1,be1,g2,be2,Wfc,bfc}, edge_index
  const float* x     = (const float*)d_in[0];
  const float* ew    = (const float*)d_in[1];
  const float* Wl1   = (const float*)d_in[2];
  const float* bl1   = (const float*)d_in[3];
  const float* Wr1   = (const float*)d_in[4];
  const float* br1   = (const float*)d_in[5];
  const float* We1   = (const float*)d_in[6];
  const float* att1  = (const float*)d_in[7];
  const float* bo1   = (const float*)d_in[8];
  const float* Wl2   = (const float*)d_in[9];
  const float* bl2   = (const float*)d_in[10];
  const float* Wr2   = (const float*)d_in[11];
  const float* br2   = (const float*)d_in[12];
  const float* We2   = (const float*)d_in[13];
  const float* att2  = (const float*)d_in[14];
  const float* bo2   = (const float*)d_in[15];
  const float* g1    = (const float*)d_in[16];
  const float* be1   = (const float*)d_in[17];
  const float* g2    = (const float*)d_in[18];
  const float* be2   = (const float*)d_in[19];
  const float* Wfc   = (const float*)d_in[20];
  const float* bfc   = (const float*)d_in[21];
  const long long* eidx = (const long long*)d_in[22];

  const long long N = in_sizes[0] / NCLS;
  const long long E = in_sizes[1];
  const long long* srcI = eidx;
  const long long* dstI = eidx + E;
  float* out = (float*)d_out;

  // Workspace carve-out
  char* ws = (char*)d_ws;
  size_t off = 0;
  auto carve = [&](size_t bytes) -> float* {
    float* p = (float*)(ws + off);
    off = (off + bytes + 255) & ~(size_t)255;
    return p;
  };
  float* XL    = carve((size_t)N * HC * 4);
  float* XR    = carve((size_t)N * HC * 4);
  float* HAGG  = carve((size_t)N * HC * 4);
  float* ALPHA = carve((size_t)(E + N) * 2 * 4);
  float* NMAX  = carve((size_t)N * 2 * 4);
  float* NDEN  = carve((size_t)N * 2 * 4);
  float* DEG   = carve((size_t)N * 4);
  float* SEW   = carve((size_t)N * 4);   // becomes loop_attr
  float* MU    = carve(HC * 4);
  float* RSIG  = carve(HC * 4);

  auto cdiv = [](long long a, long long b) -> unsigned { return (unsigned)((a + b - 1) / b); };
  const dim3 B256(256);
  const dim3 gemmGrid(cdiv(N, 64), HC / 64);
  const dim3 fcGrid(cdiv(N, 64), 1);
  const unsigned edgeBlkE = cdiv(E, 8);      // 8 waves/block, wave-per-edge
  const unsigned edgeBlkN = cdiv(N, 8);
  const unsigned ehE = cdiv(E * 2, 256);
  const unsigned ehN = cdiv(N * 2, 256);
  const unsigned nodeCh = cdiv(N * HC, 256);

  // ---- Layer 1 (K=26 -> generic path) ----
  k_gemm_wmma<false><<<gemmGrid, B256, 0, stream>>>(x, Wl1, bl1, XL, (int)N, NCLS, HC);
  k_gemm_wmma<false><<<gemmGrid, B256, 0, stream>>>(x, Wr1, br1, XR, (int)N, NCLS, HC);
  k_fill_bias<<<nodeCh, B256, 0, stream>>>(HAGG, bo1, N * HC);
  k_init_softmax<<<cdiv(N * 2, 256), B256, 0, stream>>>(NMAX, NDEN, N * 2);
  k_edge_alpha<false><<<edgeBlkE, B256, 0, stream>>>(srcI, dstI, ew, nullptr, XL, XR, We1, att1, ALPHA, E);
  k_alpha_max<false><<<ehE, B256, 0, stream>>>(dstI, ALPHA, NMAX, E);
  k_alpha_exp<false><<<ehE, B256, 0, stream>>>(dstI, ALPHA, NMAX, NDEN, E);
  k_scatter<false><<<edgeBlkE, B256, 0, stream>>>(srcI, dstI, ALPHA, NDEN, XL, HAGG, E);
  k_bn_stats<<<HC, B256, 0, stream>>>(HAGG, MU, RSIG, (int)N);
  k_bn_act_drop<<<nodeCh, B256, 0, stream>>>(HAGG, MU, RSIG, g1, be1, 1u, N * HC);

  // self-loop attrs for layer 2 (mean of incoming edge_weight per target)
  k_zero2<<<cdiv(N, 256), B256, 0, stream>>>(DEG, SEW, N);
  k_deg_sew<<<cdiv(E, 256), B256, 0, stream>>>(dstI, ew, DEG, SEW, E);
  k_loop_attr<<<cdiv(N, 256), B256, 0, stream>>>(DEG, SEW, N);

  // ---- Layer 2 (K=256 -> vectorized WMMA path; edges + self-loops) ----
  k_gemm_wmma<true><<<gemmGrid, B256, 0, stream>>>(HAGG, Wl2, bl2, XL, (int)N, HC, HC);
  k_gemm_wmma<true><<<gemmGrid, B256, 0, stream>>>(HAGG, Wr2, br2, XR, (int)N, HC, HC);
  k_fill_bias<<<nodeCh, B256, 0, stream>>>(HAGG, bo2, N * HC);   // stream-ordered: gemms already consumed h1
  k_init_softmax<<<cdiv(N * 2, 256), B256, 0, stream>>>(NMAX, NDEN, N * 2);
  float* ALPHA_SELF = ALPHA + (size_t)E * 2;
  k_edge_alpha<false><<<edgeBlkE, B256, 0, stream>>>(srcI, dstI, ew, nullptr, XL, XR, We2, att2, ALPHA, E);
  k_edge_alpha<true ><<<edgeBlkN, B256, 0, stream>>>(nullptr, nullptr, nullptr, SEW, XL, XR, We2, att2, ALPHA_SELF, N);
  k_alpha_max<false><<<ehE, B256, 0, stream>>>(dstI, ALPHA, NMAX, E);
  k_alpha_max<true ><<<ehN, B256, 0, stream>>>(nullptr, ALPHA_SELF, NMAX, N);
  k_alpha_exp<false><<<ehE, B256, 0, stream>>>(dstI, ALPHA, NMAX, NDEN, E);
  k_alpha_exp<true ><<<ehN, B256, 0, stream>>>(nullptr, ALPHA_SELF, NMAX, NDEN, N);
  k_scatter<false><<<edgeBlkE, B256, 0, stream>>>(srcI, dstI, ALPHA, NDEN, XL, HAGG, E);
  k_scatter<true ><<<edgeBlkN, B256, 0, stream>>>(nullptr, nullptr, ALPHA_SELF, NDEN, XL, HAGG, N);
  k_bn_stats<<<HC, B256, 0, stream>>>(HAGG, MU, RSIG, (int)N);
  k_bn_act_drop<<<nodeCh, B256, 0, stream>>>(HAGG, MU, RSIG, g2, be2, 2u, N * HC);

  // ---- Final FC: out = h2 @ Wfc + bfc  (N x 26, generic path) ----
  k_gemm_wmma<false><<<fcGrid, B256, 0, stream>>>(HAGG, Wfc, bfc, out, (int)N, HC, NCLS);
}
